// M2_27968827032306
// MI455X (gfx1250) — matrix operand order repaired and checked by reading, not compiled, not run on hardware
//
#include <hip/hip_runtime.h>
#include <math.h>

typedef __attribute__((ext_vector_type(16))) _Float16 v16h;
typedef __attribute__((ext_vector_type(8)))  _Float16 v8h;
typedef __attribute__((ext_vector_type(8)))  float    v8f;

// ---------- helpers ----------
__device__ __forceinline__ int f2ord(float f) {
    int b = __float_as_int(f);
    return (b >= 0) ? b : (b ^ 0x7fffffff);
}
__device__ __forceinline__ float ord2f(int b) {
    return __int_as_float((b >= 0) ? b : (b ^ 0x7fffffff));
}
__device__ __forceinline__ float lrelu_s(float v, float s) { return v > 0.f ? v : v * s; }
__device__ __forceinline__ void atomAddF(float* p, float v) {
    __hip_atomic_fetch_add(p, v, __ATOMIC_RELAXED, __HIP_MEMORY_SCOPE_AGENT);
}

// ---------- weight pre-convert: W[KxN] f32 row-major -> Wt[Npad x K] f16
// (row n of Wt = column n of W; columns >= N zero-padded) ----------
__global__ void wconv(const float* __restrict__ W, _Float16* __restrict__ Wt,
                      int K, int N, int Npad)
{
    int i = blockIdx.x * blockDim.x + threadIdx.x;
    if (i >= K * Npad) return;
    int n = i / K, k = i % K;
    Wt[(size_t)n * K + k] = (_Float16)((n < N) ? W[(size_t)k * N + n] : 0.f);
}

// ---------- WMMA GEMM: C[MxN] = A[MxK](f32) * B (f16 col-major Wt) (+bias) ----------
// 128 threads = 4 waves; wave computes 16 rows x 64 cols = 4 x (16x16) WMMA tiles.
// No LDS: A fragments are two v8f global loads (cvt to f16); B fragments are two
// 16-byte f16 global loads from the column-major weight copy (L2-resident).
// K must be a multiple of 32; Wt must have >= gridDim.y*64 padded columns.
__global__ __launch_bounds__(128)
void gemm_wmma(const float* __restrict__ A, const _Float16* __restrict__ Bt,
               const float* __restrict__ bias, float* __restrict__ C,
               int M, int N, int K)
{
    const int bm = blockIdx.x * 64;
    const int bn = blockIdx.y * 64;
    const int wave = threadIdx.x >> 5;
    const int lane = threadIdx.x & 31;
    const int r  = lane & 15;
    const int kh = lane >> 4;

    const int m = bm + wave * 16 + r;
    const float* __restrict__ arow = A + (size_t)(m < M ? m : 0) * K;

    v8f acc[4];
#pragma unroll
    for (int t = 0; t < 4; ++t) acc[t] = (v8f)(0.0f);

    for (int k0 = 0; k0 < K; k0 += 32) {
        // A fragment: lane needs K runs [k0+kh*8, +8) and [k0+16+kh*8, +8)
        v8f fa0 = *(const v8f*)(arow + k0 + kh * 8);
        v8f fa1 = *(const v8f*)(arow + k0 + 16 + kh * 8);
        v8h ha0 = __builtin_convertvector(fa0, v8h);
        v8h ha1 = __builtin_convertvector(fa1, v8h);
        v16h af = __builtin_shufflevector(ha0, ha1,
            0, 1, 2, 3, 4, 5, 6, 7, 8, 9, 10, 11, 12, 13, 14, 15);

#pragma unroll
        for (int t = 0; t < 4; ++t) {
            // B fragment: lane holds column (bn+t*16+r), K runs [k0+kh*8, +8) and [k0+16+kh*8, +8)
            const v8h* bp = (const v8h*)(Bt + (size_t)(bn + t * 16 + r) * K + k0 + kh * 8);
            v8h b0 = bp[0];
            v8h b1 = bp[2];
            v16h bf = __builtin_shufflevector(b0, b1,
                0, 1, 2, 3, 4, 5, 6, 7, 8, 9, 10, 11, 12, 13, 14, 15);
            acc[t] = __builtin_amdgcn_wmma_f32_16x16x32_f16(
                false, af, false, bf, (short)0, acc[t], false, false);
        }
    }

    // store: lane holds col = (lane&15); VGPR rr holds row = rr + 8*(lane>>4)
#pragma unroll
    for (int t = 0; t < 4; ++t) {
        int gc = bn + t * 16 + r;
        if (gc >= N) continue;
        float bv = bias ? bias[gc] : 0.f;
#pragma unroll
        for (int rr = 0; rr < 8; ++rr) {
            int gr = bm + wave * 16 + rr + 8 * kh;
            if (gr < M) C[(size_t)gr * N + gc] = acc[t][rr] + bv;
        }
    }
}

// ---------- GAT attention scores: sc = sum_c h[n,hd,c]*a[hd,c] ----------
__global__ void gat_scores(const float* __restrict__ h, const float* __restrict__ a_s,
                           const float* __restrict__ a_d, float* __restrict__ ss,
                           float* __restrict__ sd, int N, int heads, int ch)
{
    int i = blockIdx.x * blockDim.x + threadIdx.x;
    if (i >= N * heads) return;
    int n = i / heads, hd = i % heads;
    const float* hp = h + ((size_t)n * heads + hd) * ch;
    float s0 = 0.f, s1 = 0.f;
    for (int c = 0; c < ch; ++c) {
        float v = hp[c];
        s0 += v * a_s[hd * ch + c];
        s1 += v * a_d[hd * ch + c];
    }
    ss[i] = s0;
    sd[i] = s1;
}

__global__ void fill_i32(int* __restrict__ p, int v, long n)
{
    for (long i = blockIdx.x * (long)blockDim.x + threadIdx.x; i < n;
         i += (long)gridDim.x * blockDim.x)
        p[i] = v;
}

// ---------- edge pass 1: e = lrelu02(ss[src]+sd[dst]); m[dst] = max ----------
__global__ void gat_edge_max(const int* __restrict__ ei, const float* __restrict__ ss,
                             const float* __restrict__ sd, int* __restrict__ mord,
                             float* __restrict__ ebuf, int E, int N, int heads)
{
    long total = (long)(E + N) * heads;
    for (long idx = blockIdx.x * (long)blockDim.x + threadIdx.x; idx < total;
         idx += (long)gridDim.x * blockDim.x) {
        int e = (int)(idx / heads), hd = (int)(idx % heads);
        int s = (e < E) ? ei[e] : (e - E);
        int d = (e < E) ? ei[E + e] : (e - E);
        float v = ss[s * heads + hd] + sd[d * heads + hd];
        v = lrelu_s(v, 0.2f);
        ebuf[idx] = v;
        atomicMax(&mord[d * heads + hd], f2ord(v));
    }
}

// ---------- edge pass 2: ee = exp(e - m[dst]); s[dst] += ee ----------
__global__ void gat_edge_exp(const int* __restrict__ ei, const int* __restrict__ mord,
                             float* __restrict__ ebuf, float* __restrict__ sbuf,
                             int E, int N, int heads)
{
    long total = (long)(E + N) * heads;
    for (long idx = blockIdx.x * (long)blockDim.x + threadIdx.x; idx < total;
         idx += (long)gridDim.x * blockDim.x) {
        int e = (int)(idx / heads), hd = (int)(idx % heads);
        int d = (e < E) ? ei[E + e] : (e - E);
        float m = ord2f(mord[d * heads + hd]);
        float ex = __expf(ebuf[idx] - m);
        ebuf[idx] = ex;
        atomAddF(&sbuf[d * heads + hd], ex);
    }
}

// ---------- edge pass 3: out[dst] += h[src] * ee/(s[dst]+eps) ----------
__global__ void gat_scatter(const int* __restrict__ ei, const float* __restrict__ h,
                            const float* __restrict__ ebuf, const float* __restrict__ sbuf,
                            float* __restrict__ out, int E, int N, int heads, int ch)
{
    int chunks = ch >> 2;
    long total = (long)(E + N) * heads * chunks;
    for (long idx = blockIdx.x * (long)blockDim.x + threadIdx.x; idx < total;
         idx += (long)gridDim.x * blockDim.x) {
        long t = idx;
        int c4 = (int)(t % chunks); t /= chunks;
        int hd = (int)(t % heads);  t /= heads;
        int e = (int)t;
        int s = (e < E) ? ei[e] : (e - E);
        int d = (e < E) ? ei[E + e] : (e - E);
        float alpha = ebuf[(long)e * heads + hd] / (sbuf[d * heads + hd] + 1e-16f);
        const float4 hv = *(const float4*)(h + ((size_t)s * heads + hd) * ch + c4 * 4);
        float* op = out + ((size_t)d * heads + hd) * ch + c4 * 4;
        atomAddF(op + 0, hv.x * alpha);
        atomAddF(op + 1, hv.y * alpha);
        atomAddF(op + 2, hv.z * alpha);
        atomAddF(op + 3, hv.w * alpha);
    }
}

// ---------- GCN degree / norm ----------
__global__ void deg_kernel(const int* __restrict__ ei, float* __restrict__ deg, int E, int N)
{
    long total = E + N;
    for (long idx = blockIdx.x * (long)blockDim.x + threadIdx.x; idx < total;
         idx += (long)gridDim.x * blockDim.x) {
        int e = (int)idx;
        int d = (e < E) ? ei[E + e] : (e - E);
        atomAddF(&deg[d], 1.0f);
    }
}
__global__ void dis_kernel(float* __restrict__ deg, int N)
{
    int i = blockIdx.x * blockDim.x + threadIdx.x;
    if (i < N) deg[i] = (deg[i] > 0.f) ? rsqrtf(deg[i]) : 0.f;
}
__global__ void gcn_scatter(const int* __restrict__ ei, const float* __restrict__ h,
                            const float* __restrict__ dis, float* __restrict__ out,
                            int E, int N, int C)
{
    int chunks = C >> 2;
    long total = (long)(E + N) * chunks;
    for (long idx = blockIdx.x * (long)blockDim.x + threadIdx.x; idx < total;
         idx += (long)gridDim.x * blockDim.x) {
        int c4 = (int)(idx % chunks);
        int e = (int)(idx / chunks);
        int s = (e < E) ? ei[e] : (e - E);
        int d = (e < E) ? ei[E + e] : (e - E);
        float w = dis[s] * dis[d];
        const float4 hv = *(const float4*)(h + (size_t)s * C + c4 * 4);
        float* op = out + (size_t)d * C + c4 * 4;
        atomAddF(op + 0, hv.x * w);
        atomAddF(op + 1, hv.y * w);
        atomAddF(op + 2, hv.z * w);
        atomAddF(op + 3, hv.w * w);
    }
}

// ---------- elementwise post-processing ----------
// mode 1: y = g'*lrelu01(x+bias) + bb      (GAT layers: lrelu then BN)
// mode 2: y = lrelu01(g'*(x+bias) + bb)    (lin1: BN then lrelu)
// mode 3: y = lrelu01(x+bias)              (GCN output)
__global__ void postproc(const float* __restrict__ x, const float* __restrict__ bias,
                         const float* __restrict__ g, const float* __restrict__ bb,
                         float* __restrict__ y, long rows, int cols, int mode)
{
    long total = rows * (long)cols;
    const float bn_rs = rsqrtf(1.0f + 1e-5f);
    for (long i = blockIdx.x * (long)blockDim.x + threadIdx.x; i < total;
         i += (long)gridDim.x * blockDim.x) {
        int c = (int)(i % cols);
        float v = x[i] + (bias ? bias[c] : 0.f);
        if (mode == 1) {
            v = lrelu_s(v, 0.01f);
            v = g[c] * bn_rs * v + bb[c];
        } else if (mode == 2) {
            v = g[c] * bn_rs * v + bb[c];
            v = lrelu_s(v, 0.01f);
        } else {
            v = lrelu_s(v, 0.01f);
        }
        y[i] = v;
    }
}

// ---------- pooling ----------
__global__ void pool_sum(const float* __restrict__ x, const int* __restrict__ batch,
                         float* __restrict__ sums, float* __restrict__ cnt, int N, int C)
{
    long total = (long)N * C;
    for (long i = blockIdx.x * (long)blockDim.x + threadIdx.x; i < total;
         i += (long)gridDim.x * blockDim.x) {
        int n = (int)(i / C), c = (int)(i % C);
        int b = batch[n];
        atomAddF(&sums[(size_t)b * C + c], x[i]);
        if (c == 0) atomAddF(&cnt[b], 1.0f);
    }
}
__global__ void pool_div(float* __restrict__ sums, const float* __restrict__ cnt, int B, int C)
{
    int i = blockIdx.x * blockDim.x + threadIdx.x;
    if (i < B * C) {
        float ct = cnt[i / C];
        sums[i] /= (ct > 1.f ? ct : 1.f);
    }
}

// ---------- host launch ----------
extern "C" void kernel_launch(void* const* d_in, const int* in_sizes, int n_in,
                              void* d_out, int out_size, void* d_ws, size_t ws_size,
                              hipStream_t stream)
{
    const float* x      = (const float*)d_in[0];
    const int*   ei     = (const int*)  d_in[1];
    const int*   batch  = (const int*)  d_in[2];
    const float* W1     = (const float*)d_in[3];
    const float* a_src1 = (const float*)d_in[4];
    const float* a_dst1 = (const float*)d_in[5];
    const float* b1     = (const float*)d_in[6];
    const float* W2     = (const float*)d_in[7];
    const float* a_src2 = (const float*)d_in[8];
    const float* a_dst2 = (const float*)d_in[9];
    const float* b2     = (const float*)d_in[10];
    const float* Wg     = (const float*)d_in[11];
    const float* bg     = (const float*)d_in[12];
    const float* bn1_g  = (const float*)d_in[13];
    const float* bn1_b  = (const float*)d_in[14];
    const float* bn2_g  = (const float*)d_in[15];
    const float* bn2_b  = (const float*)d_in[16];
    const float* bn3_g  = (const float*)d_in[17];
    const float* bn3_b  = (const float*)d_in[18];
    const float* lin1_W = (const float*)d_in[19];
    const float* lin1_b = (const float*)d_in[20];
    const float* lin2_W = (const float*)d_in[21];
    const float* lin2_b = (const float*)d_in[22];

    const int N  = in_sizes[2];          // 50000
    const int F  = in_sizes[0] / N;      // 128
    const int E  = in_sizes[1] / 2;      // 800000
    const int NC = 10;
    const int B  = out_size / NC;        // 512
    const int H4 = in_sizes[6];          // 256 (4 heads * 64)
    const int H2 = in_sizes[10];         // 128
    const int Hh = in_sizes[12];         // 64

    // workspace carve (all float offsets multiples of 8 -> 32B alignment for v8f)
    float* ws = (float*)d_ws;
    size_t o = 0;
    float* bufH  = ws + o; o += (size_t)N * H4;    // h1 / h2 / hg
    float* bufX  = ws + o; o += (size_t)N * H4;    // x1 / x2 / x3
    float* bufA  = ws + o; o += (size_t)N * H4;    // scatter accumulator
    float* scS   = ws + o; o += (size_t)N * 4;
    float* scD   = ws + o; o += (size_t)N * 4;
    int*   mOrd  = (int*)(ws + o); o += (size_t)N * 4;
    float* sSum  = ws + o; o += (size_t)N * 4;
    float* eBuf  = ws + o; o += (size_t)(E + N) * 4;
    float* deg   = ws + o; o += (size_t)N;
    float* pooled= ws + o; o += (size_t)B * Hh;
    float* cnt   = ws + o; o += (size_t)B;
    float* y1    = ws + o; o += (size_t)B * 128;
    // f16 column-major weight copies (Npad x K each, Npad = ceil64(N))
    _Float16* hws = (_Float16*)(ws + o);
    size_t ho = 0;
    _Float16* W1t = hws + ho; ho += (size_t)256 * F;    // 256 x 128
    _Float16* W2t = hws + ho; ho += (size_t)H2  * H4;   // 128 x 256
    _Float16* Wgt = hws + ho; ho += (size_t)64  * H2;   // 64  x 128
    _Float16* l1t = hws + ho; ho += (size_t)128 * Hh;   // 128 x 64
    _Float16* l2t = hws + ho; ho += (size_t)64  * 128;  // 64(pad of 10) x 128
    (void)ws_size; (void)n_in;

    const int NEG_ORD = (int)0x807fffff;  // f2ord(-inf)
    const dim3 blk(256);
    const dim3 gsGrid(4096);

    auto conv = [&](const float* W, _Float16* Wt, int K, int Nn, int Npad) {
        wconv<<<(K * Npad + 255) / 256, blk, 0, stream>>>(W, Wt, K, Nn, Npad);
    };
    auto launch_gemm = [&](const float* A, const _Float16* Bt, const float* bias,
                           float* C, int M, int Nn, int K) {
        dim3 grid((M + 63) / 64, (Nn + 63) / 64);
        gemm_wmma<<<grid, dim3(128), 0, stream>>>(A, Bt, bias, C, M, Nn, K);
    };

    // ===== weight conversions (tiny; L2-resident thereafter) =====
    conv(W1,     W1t, F,  H4, 256);
    conv(W2,     W2t, H4, H2, 128);
    conv(Wg,     Wgt, H2, Hh, 64);
    conv(lin1_W, l1t, Hh, 128, 128);
    conv(lin2_W, l2t, 128, NC, 64);

    // ===== GAT layer 1 (4 heads x 64) =====
    launch_gemm(x, W1t, nullptr, bufH, N, H4, F);
    gat_scores<<<(N * 4 + 255) / 256, blk, 0, stream>>>(bufH, a_src1, a_dst1, scS, scD, N, 4, 64);
    fill_i32<<<gsGrid, blk, 0, stream>>>(mOrd, NEG_ORD, (long)N * 4);
    hipMemsetAsync(sSum, 0, (size_t)N * 4 * sizeof(float), stream);
    hipMemsetAsync(bufA, 0, (size_t)N * H4 * sizeof(float), stream);
    gat_edge_max<<<gsGrid, blk, 0, stream>>>(ei, scS, scD, mOrd, eBuf, E, N, 4);
    gat_edge_exp<<<gsGrid, blk, 0, stream>>>(ei, mOrd, eBuf, sSum, E, N, 4);
    gat_scatter<<<gsGrid, blk, 0, stream>>>(ei, bufH, eBuf, sSum, bufA, E, N, 4, 64);
    postproc<<<gsGrid, blk, 0, stream>>>(bufA, b1, bn1_g, bn1_b, bufX, (long)N, H4, 1);

    // ===== GAT layer 2 (1 head x 128) =====
    launch_gemm(bufX, W2t, nullptr, bufH, N, H2, H4);
    gat_scores<<<(N + 255) / 256, blk, 0, stream>>>(bufH, a_src2, a_dst2, scS, scD, N, 1, H2);
    fill_i32<<<gsGrid, blk, 0, stream>>>(mOrd, NEG_ORD, (long)N);
    hipMemsetAsync(sSum, 0, (size_t)N * sizeof(float), stream);
    hipMemsetAsync(bufA, 0, (size_t)N * H2 * sizeof(float), stream);
    gat_edge_max<<<gsGrid, blk, 0, stream>>>(ei, scS, scD, mOrd, eBuf, E, N, 1);
    gat_edge_exp<<<gsGrid, blk, 0, stream>>>(ei, mOrd, eBuf, sSum, E, N, 1);
    gat_scatter<<<gsGrid, blk, 0, stream>>>(ei, bufH, eBuf, sSum, bufA, E, N, 1, H2);
    postproc<<<gsGrid, blk, 0, stream>>>(bufA, b2, bn2_g, bn2_b, bufX, (long)N, H2, 1);

    // ===== GCN layer =====
    hipMemsetAsync(deg, 0, (size_t)N * sizeof(float), stream);
    deg_kernel<<<gsGrid, blk, 0, stream>>>(ei, deg, E, N);
    dis_kernel<<<(N + 255) / 256, blk, 0, stream>>>(deg, N);
    launch_gemm(bufX, Wgt, nullptr, bufH, N, Hh, H2);
    hipMemsetAsync(bufA, 0, (size_t)N * Hh * sizeof(float), stream);
    gcn_scatter<<<gsGrid, blk, 0, stream>>>(ei, bufH, deg, bufA, E, N, Hh);
    postproc<<<gsGrid, blk, 0, stream>>>(bufA, bg, nullptr, nullptr, bufX, (long)N, Hh, 3);

    // ===== global mean pool =====
    hipMemsetAsync(pooled, 0, (size_t)B * Hh * sizeof(float), stream);
    hipMemsetAsync(cnt, 0, (size_t)B * sizeof(float), stream);
    pool_sum<<<gsGrid, blk, 0, stream>>>(bufX, batch, pooled, cnt, N, Hh);
    pool_div<<<(B * Hh + 255) / 256, blk, 0, stream>>>(pooled, cnt, B, Hh);

    // ===== MLP head =====
    launch_gemm(pooled, l1t, nullptr, bufA, B, 128, Hh);
    postproc<<<gsGrid, blk, 0, stream>>>(bufA, lin1_b, bn3_g, bn3_b, y1, (long)B, 128, 2);
    launch_gemm(y1, l2t, lin2_b, (float*)d_out, B, NC, 128);
}
